// InteractionNetwork_90847148245379
// MI455X (gfx1250) — matrix operand order repaired and checked by reading, not compiled, not run on hardware
//
#include <hip/hip_runtime.h>

typedef __bf16 bf16_t;
typedef __attribute__((ext_vector_type(16))) __bf16 v16bf;
typedef __attribute__((ext_vector_type(8)))  __bf16 v8bf;
typedef __attribute__((ext_vector_type(8)))  float  v8f;

#define B_SZ   64
#define N_OBJ  100
#define R_REL  9900
#define RP     9984    // R padded to multiple of 128
#define K1     32      // relational layer-1 K: 11 real + bias row at k=11
#define H_REL  160     // 150 -> 160, unit column at n=150
#define E_PAD  112     // 100 -> 112
#define KO     128     // object MLP K pad; bias rows at k=105 / k=100
#define NO     112     // object MLP hidden pad, unit column at n=100
#define KS_R   (RP / 32)   // 312 k-steps in scatter

// ---------- WMMA fragment helpers (CDNA5 16x16x32 bf16, wave32) ----------
// A 16x32 (MxK): lane<16 -> row m=lane, K={0..7,16..23}; lane>=16 -> K={8..15,24..31}
// laneBase = src + (lane&15)*ld + (lane>>4)*8 ; koff is a compile-time constant,
// so both 16-B loads fold into the DS/FLAT offset field (no per-access VALU).
__device__ __forceinline__ v16bf a_frag(const bf16_t* laneBase, int koff) {
  v8bf lo = *(const v8bf*)(laneBase + koff);
  v8bf hi = *(const v8bf*)(laneBase + koff + 16);
  return __builtin_shufflevector(lo, hi, 0, 1, 2, 3, 4, 5, 6, 7,
                                 8, 9, 10, 11, 12, 13, 14, 15);
}

// B 32x16 (KxN) fragment-major ("swizzled") tile of 512 bf16:
// element (k,n) at lane*16 + (k/16)*8 + k%8 with lane = ((k/8)%2)*16 + n.
// laneBase = arr + lane*16 ; toff (tile*512) is a compile-time constant.
__device__ __forceinline__ v16bf b_frag(const bf16_t* laneBase, int toff) {
  v8bf lo = *(const v8bf*)(laneBase + toff);
  v8bf hi = *(const v8bf*)(laneBase + toff + 8);
  return __builtin_shufflevector(lo, hi, 0, 1, 2, 3, 4, 5, 6, 7,
                                 8, 9, 10, 11, 12, 13, 14, 15);
}

__device__ __forceinline__ v8f wmma_bf16(v16bf a, v16bf b, v8f c) {
  return __builtin_amdgcn_wmma_f32_16x16x32_bf16(false, a, false, b, (short)0, c,
                                                 false, false);
}

__device__ __forceinline__ float relu(float x) {
  return __builtin_amdgcn_fmed3f(x, 0.f, __builtin_inff());
}

// Swizzle a row-major f32 [srcK x srcN] weight into bf16 fragment-major tiles,
// bias folded in as weight row `biasRow`, optional unit column at
// (biasRow, unitCol) to propagate the constant-1 input forward.
__device__ __forceinline__ void swizzleB(const float* src, int srcK, int srcN,
                                         const float* bias, int biasN, int biasRow,
                                         int unitCol,
                                         bf16_t* dst, int ksteps, int ntc, int t) {
  int total = ksteps * ntc * 512;
  for (int i = t; i < total; i += 256) {
    int tile   = i >> 9;
    int within = i & 511;
    int lane   = within >> 4;
    int elem   = within & 15;
    int ks = tile / ntc, nt = tile % ntc;
    int k = ks * 32 + (elem >> 3) * 16 + (lane >> 4) * 8 + (elem & 7);
    int n = nt * 16 + (lane & 15);
    float v = 0.f;
    if (k < srcK && n < srcN)        v = src[k * srcN + n];
    else if (k == biasRow) {
      if (n < biasN)                 v = bias[n];
      else if (n == unitCol)         v = 1.f;
    }
    dst[i] = (bf16_t)v;
  }
}

// ---------- kernel 1: weight pad/convert/swizzle (bias folded) ----------
__global__ void prep_kernel(const float* rW1, const float* rb1, const float* rW2,
                            const float* rb2, const float* rW3, const float* rb3,
                            const float* oW1, const float* ob1, const float* oW2,
                            const float* ob2, const float* oW3, const float* ob3,
                            bf16_t* W1p, bf16_t* W2p, bf16_t* W3p,
                            bf16_t* U1p, bf16_t* U2p, bf16_t* U3p) {
  int t = threadIdx.x;
  swizzleB(rW1, 11, 150, rb1, 150, 11, 150, W1p, 1, 10, t);   // 32x160
  swizzleB(rW2, 150, 150, rb2, 150, 150, 150, W2p, 5, 10, t); // 160x160
  swizzleB(rW3, 150, 100, rb3, 100, 150, -1, W3p, 5, 7, t);   // 160x112
  swizzleB(oW1, 105, 100, ob1, 100, 105, 100, U1p, 4, 7, t);  // 128x112
  swizzleB(oW2, 100, 100, ob2, 100, 100, 100, U2p, 4, 7, t);  // 128x112
  swizzleB(oW3, 100, 2,   ob3, 2,   100, -1, U3p, 4, 1, t);   // 128x16
}

// ---------- kernel 2: gather (S^T O, R^T O, rel_info, 1) -> bf16 X1 [B,RP,32] ----------
__global__ void __launch_bounds__(256) gather_kernel(const float* objects,
                                                     const float* S, const float* Rr,
                                                     const float* relinfo, bf16_t* X1) {
  __shared__ float sO[N_OBJ * 5];
  int b = blockIdx.y;
  int r = blockIdx.x * 256 + threadIdx.x;   // r < RP by grid construction
  for (int i = threadIdx.x; i < N_OBJ * 5; i += 256)
    sO[i] = objects[(size_t)b * (N_OBJ * 5) + i];
  __syncthreads();

  float acc[10];
#pragma unroll
  for (int d = 0; d < 10; ++d) acc[d] = 0.f;
  float ri = 0.f, one = 0.f;
  if (r < R_REL) {
    const float* Sp = S  + (size_t)b * N_OBJ * R_REL + r;
    const float* Rp = Rr + (size_t)b * N_OBJ * R_REL + r;
    for (int n = 0; n < N_OBJ; ++n) {
      float sv = Sp[(size_t)n * R_REL];
      float rv = Rp[(size_t)n * R_REL];
#pragma unroll
      for (int d = 0; d < 5; ++d) {
        float o = sO[n*5 + d];
        acc[d]     += sv * o;
        acc[5 + d] += rv * o;
      }
    }
    ri  = relinfo[(size_t)b * R_REL + r];
    one = 1.f;                      // bias input for the folded bias row
  }
  union { bf16_t h[32]; uint4 v4[4]; } u;
#pragma unroll
  for (int c = 0; c < 32; ++c) u.h[c] = (bf16_t)0.f;
#pragma unroll
  for (int d = 0; d < 10; ++d) u.h[d] = (bf16_t)acc[d];
  u.h[10] = (bf16_t)ri;
  u.h[11] = (bf16_t)one;
  uint4* dst = (uint4*)(X1 + ((size_t)b * RP + r) * K1);
#pragma unroll
  for (int j = 0; j < 4; ++j) dst[j] = u.v4[j];
}

// ---------- kernel 3: relational MLP 32->160->160->112, WMMA ----------
__global__ void __launch_bounds__(256) relmlp_kernel(const bf16_t* X1,
    const bf16_t* W1p, const bf16_t* W2p, const bf16_t* W3p, bf16_t* Esw) {
  __shared__ alignas(16) bf16_t sW1[1 * 10 * 512];
  __shared__ alignas(16) bf16_t sW2[5 * 10 * 512];
  __shared__ alignas(16) bf16_t sW3[5 * 7 * 512];
  __shared__ alignas(16) bf16_t sH[8][16 * H_REL];   // per-wave private H staging

  int t = threadIdx.x;
  { // wide LDS staging of swizzled weights
    const uint4* s1 = (const uint4*)W1p; uint4* d1 = (uint4*)sW1;
    for (int i = t; i < (1*10*512)/8; i += 256) d1[i] = s1[i];
    const uint4* s2 = (const uint4*)W2p; uint4* d2 = (uint4*)sW2;
    for (int i = t; i < (5*10*512)/8; i += 256) d2[i] = s2[i];
    const uint4* s3 = (const uint4*)W3p; uint4* d3 = (uint4*)sW3;
    for (int i = t; i < (5*7*512)/8; i += 256) d3[i] = s3[i];
  }
  __syncthreads();

  int w    = t >> 5;
  int lane = t & 31;
  int hf   = lane >> 4;
  int nlo  = lane & 15;
  int m    = lane & 15;
  int kb8  = hf * 8;
  int b    = blockIdx.y;
  int r0   = blockIdx.x * 128 + w * 16;
  bf16_t* Hw = sH[w];
  const v8f vzero = {};

  // lane-adjusted bases: all fragment accesses become base + const-offset
  const bf16_t* w1b = sW1 + lane * 16;
  const bf16_t* w2b = sW2 + lane * 16;
  const bf16_t* w3b = sW3 + lane * 16;
  const bf16_t* hA  = Hw + m * H_REL + kb8;        // A-frag base into H
  bf16_t*       hS  = Hw + (hf * 8) * H_REL + nlo; // D-store base into H

  // layer 1: [16x32] @ [32x160], bias folded, relu -> Hw
  {
    v16bf a1 = a_frag(X1 + ((size_t)b * RP + r0) * K1 + m * K1 + kb8, 0);
#pragma unroll
    for (int nt = 0; nt < 10; ++nt) {
      v8f c = wmma_bf16(a1, b_frag(w1b, nt * 512), vzero);   // SRC2 = inline 0
#pragma unroll
      for (int rr = 0; rr < 8; ++rr)
        hS[rr * H_REL + nt * 16] = (bf16_t)relu(c[rr]);
    }
  }
  // layer 2: [16x160] @ [160x160], bias via unit column, relu -> Hw
  {
    v8f acc[10];
    {
      v16bf a = a_frag(hA, 0);
#pragma unroll
      for (int nt = 0; nt < 10; ++nt)
        acc[nt] = wmma_bf16(a, b_frag(w2b, nt * 512), vzero);
    }
#pragma unroll
    for (int ks = 1; ks < 5; ++ks) {
      v16bf a = a_frag(hA, ks * 32);
#pragma unroll
      for (int nt = 0; nt < 10; ++nt)
        acc[nt] = wmma_bf16(a, b_frag(w2b, (ks * 10 + nt) * 512), acc[nt]);
    }
#pragma unroll
    for (int nt = 0; nt < 10; ++nt)
#pragma unroll
      for (int rr = 0; rr < 8; ++rr)
        hS[rr * H_REL + nt * 16] = (bf16_t)relu(acc[nt][rr]);
  }
  // layer 3: [16x160] @ [160x112], relu -> Esw (B-swizzled, b128 stores)
  {
    v8f acc[7];
    {
      v16bf a = a_frag(hA, 0);
#pragma unroll
      for (int nt = 0; nt < 7; ++nt)
        acc[nt] = wmma_bf16(a, b_frag(w3b, nt * 512), vzero);
    }
#pragma unroll
    for (int ks = 1; ks < 5; ++ks) {
      v16bf a = a_frag(hA, ks * 32);
#pragma unroll
      for (int nt = 0; nt < 7; ++nt)
        acc[nt] = wmma_bf16(a, b_frag(w3b, (ks * 7 + nt) * 512), acc[nt]);
    }
    int ksr = r0 >> 5;            // k-step (r tile of 32) this wave contributes to
    int h2  = (r0 >> 4) & 1;      // which k-half of the B tile
#pragma unroll
    for (int nt = 0; nt < 7; ++nt) {
      union { bf16_t h[8]; uint4 q; } u;
#pragma unroll
      for (int rr = 0; rr < 8; ++rr) u.h[rr] = (bf16_t)relu(acc[nt][rr]);
      size_t tbase = (((size_t)b * KS_R + ksr) * 7 + nt) * 512
                   + (hf * 16 + nlo) * 16 + h2 * 8;
      *(uint4*)(Esw + tbase) = u.q;
    }
  }
}

// ---------- kernel 4: scatter  ER[b] = Rr[b] (100x9900) @ E[b] (9900x100), WMMA ----------
__global__ void __launch_bounds__(256) scatter_kernel(const float* Rr, const bf16_t* Esw,
                                                      float* ER) {
  __shared__ alignas(16) bf16_t sA[E_PAD * 32];   // 112 n-rows x 32 k (Rr, f32->bf16)
  __shared__ alignas(16) bf16_t sB[7 * 512];      // 7 swizzled E tiles for this k-step
  int b    = blockIdx.x;
  int t    = threadIdx.x;
  int w    = t >> 5;
  int lane = t & 31;
  int hf   = lane >> 4;
  int nlo  = lane & 15;
  const bf16_t* aB = sA + (lane & 15) * 32 + hf * 8;  // A-frag lane base
  const bf16_t* bB = sB + lane * 16;                  // B-frag lane base

  v8f acc[7];
#pragma unroll
  for (int s = 0; s < 7; ++s)
#pragma unroll
    for (int rr = 0; rr < 8; ++rr) acc[s][rr] = 0.f;

  for (int ks = 0; ks < KS_R; ++ks) {
    int rbase = ks * 32;
    // Rr panel: packed bf16 pair writes (b32 stores), coalesced f32 pair reads
    for (int i = t; i < E_PAD * 16; i += 256) {
      int n  = i >> 4;
      int kp = (i & 15) * 2;
      int r  = rbase + kp;
      float v0 = 0.f, v1 = 0.f;
      if (n < N_OBJ && r < R_REL) {
        const float* p = Rr + ((size_t)b * N_OBJ + n) * R_REL + r;
        v0 = p[0];
        v1 = p[1];
        if (r + 32 < R_REL) __builtin_prefetch(p + 32, 0, 0);  // next k-panel
      }
      union { bf16_t h[2]; unsigned int u32; } pk;
      pk.h[0] = (bf16_t)v0; pk.h[1] = (bf16_t)v1;
      *(unsigned int*)(sA + n * 32 + kp) = pk.u32;
    }
    // E panel: verbatim uint4 copy of 7 pre-swizzled tiles
    {
      const uint4* src = (const uint4*)(Esw + (((size_t)b * KS_R + ks) * 7) * 512);
      uint4* dst = (uint4*)sB;
      for (int i = t; i < (7 * 512) / 8; i += 256) dst[i] = src[i];
    }
    __syncthreads();
#pragma unroll
    for (int s = 0; s < 7; ++s) {
      int p = w + 8 * s;                 // wave-uniform -> EXEC stays full
      if (p < 49) {
        int mt = p / 7, nt = p % 7;
        acc[s] = wmma_bf16(a_frag(aB, mt * 512), b_frag(bB, nt * 512), acc[s]);
      }
    }
    __syncthreads();
  }
#pragma unroll
  for (int s = 0; s < 7; ++s) {
    int p = w + 8 * s;
    if (p < 49) {
      int mt = p / 7, nt = p % 7;
#pragma unroll
      for (int rr = 0; rr < 8; ++rr)
        ER[((size_t)b * E_PAD + mt*16 + hf*8 + rr) * E_PAD + nt*16 + nlo] = acc[s][rr];
    }
  }
}

// ---------- kernel 5: object MLP 128->112->112->16(2), WMMA ----------
__device__ __forceinline__ float obj_in_fetch(const float* objects, const float* ER,
                                              int b, int n, int k) {
  if (n >= N_OBJ) return 0.f;
  if (k < 5)    return objects[((size_t)b * N_OBJ + n) * 5 + k];
  if (k < 105)  return ER[((size_t)b * E_PAD + n) * E_PAD + (k - 5)];
  if (k == 105) return 1.f;     // bias input for the folded bias row
  return 0.f;
}

__global__ void __launch_bounds__(256) objmlp_kernel(const float* objects, const float* ER,
    const bf16_t* U1p, const bf16_t* U2p, const bf16_t* U3p, float* out) {
  __shared__ alignas(16) bf16_t sX[NO * KO];      // obj_in: 112 rows x 128 k
  __shared__ alignas(16) bf16_t sU1[4 * 7 * 512];
  __shared__ alignas(16) bf16_t sU2[4 * 7 * 512];
  __shared__ alignas(16) bf16_t sU3[4 * 1 * 512];
  __shared__ alignas(16) bf16_t sH[7][16 * KO];

  int b = blockIdx.x;
  int t = threadIdx.x;
  // obj_in: packed bf16 pair writes
  for (int i = t; i < (NO * KO) / 2; i += 256) {
    int n  = i >> 6;          // 64 pairs per row
    int kp = (i & 63) * 2;
    union { bf16_t h[2]; unsigned int u32; } pk;
    pk.h[0] = (bf16_t)obj_in_fetch(objects, ER, b, n, kp);
    pk.h[1] = (bf16_t)obj_in_fetch(objects, ER, b, n, kp + 1);
    *(unsigned int*)(sX + n * KO + kp) = pk.u32;
  }
  { // wide weight staging
    const uint4* s1 = (const uint4*)U1p; uint4* d1 = (uint4*)sU1;
    const uint4* s2 = (const uint4*)U2p; uint4* d2 = (uint4*)sU2;
    for (int i = t; i < (4*7*512)/8; i += 256) { d1[i] = s1[i]; d2[i] = s2[i]; }
    const uint4* s3 = (const uint4*)U3p; uint4* d3 = (uint4*)sU3;
    for (int i = t; i < (4*512)/8; i += 256) d3[i] = s3[i];
  }
  { bf16_t* flat = &sH[0][0];
    for (int i = t; i < 7 * 16 * KO; i += 256) flat[i] = (bf16_t)0.f; }
  __syncthreads();

  int w = t >> 5;
  if (w < 7) {                         // wave-uniform
    int lane = t & 31;
    int hf   = lane >> 4;
    int nlo  = lane & 15;
    int m    = lane & 15;
    int kb8  = hf * 8;
    int mt   = w;
    bf16_t* Hw = sH[w];
    const v8f vzero = {};
    const bf16_t* u1b = sU1 + lane * 16;
    const bf16_t* u2b = sU2 + lane * 16;
    const bf16_t* u3b = sU3 + lane * 16;
    const bf16_t* xA  = sX + (mt * 16 + m) * KO + kb8;
    const bf16_t* hA  = Hw + m * KO + kb8;
    bf16_t*       hS  = Hw + (hf * 8) * KO + nlo;
    // layer 1
    {
      v8f acc[7];
      {
        v16bf a = a_frag(xA, 0);
#pragma unroll
        for (int nt = 0; nt < 7; ++nt)
          acc[nt] = wmma_bf16(a, b_frag(u1b, nt * 512), vzero);
      }
#pragma unroll
      for (int ks = 1; ks < 4; ++ks) {
        v16bf a = a_frag(xA, ks * 32);
#pragma unroll
        for (int nt = 0; nt < 7; ++nt)
          acc[nt] = wmma_bf16(a, b_frag(u1b, (ks * 7 + nt) * 512), acc[nt]);
      }
#pragma unroll
      for (int nt = 0; nt < 7; ++nt)
#pragma unroll
        for (int rr = 0; rr < 8; ++rr)
          hS[rr * KO + nt * 16] = (bf16_t)relu(acc[nt][rr]);
    }
    // layer 2
    {
      v8f acc[7];
      {
        v16bf a = a_frag(hA, 0);
#pragma unroll
        for (int nt = 0; nt < 7; ++nt)
          acc[nt] = wmma_bf16(a, b_frag(u2b, nt * 512), vzero);
      }
#pragma unroll
      for (int ks = 1; ks < 4; ++ks) {
        v16bf a = a_frag(hA, ks * 32);
#pragma unroll
        for (int nt = 0; nt < 7; ++nt)
          acc[nt] = wmma_bf16(a, b_frag(u2b, (ks * 7 + nt) * 512), acc[nt]);
      }
#pragma unroll
      for (int nt = 0; nt < 7; ++nt)
#pragma unroll
        for (int rr = 0; rr < 8; ++rr)
          hS[rr * KO + nt * 16] = (bf16_t)relu(acc[nt][rr]);
    }
    // layer 3 (no relu), N = 16 (cols 0..1 valid)
    {
      v8f acc = wmma_bf16(a_frag(hA, 0), b_frag(u3b, 0), vzero);
#pragma unroll
      for (int ks = 1; ks < 4; ++ks)
        acc = wmma_bf16(a_frag(hA, ks * 32), b_frag(u3b, ks * 512), acc);
#pragma unroll
      for (int rr = 0; rr < 8; ++rr) {
        int row = mt*16 + hf*8 + rr;
        if (row < N_OBJ && nlo < 2)
          out[((size_t)b * N_OBJ + row) * 2 + nlo] = acc[rr];
      }
    }
  }
}

// ---------- host launcher ----------
extern "C" void kernel_launch(void* const* d_in, const int* in_sizes, int n_in,
                              void* d_out, int out_size, void* d_ws, size_t ws_size,
                              hipStream_t stream) {
  const float* objects = (const float*)d_in[0];
  const float* S       = (const float*)d_in[1];
  const float* Rr      = (const float*)d_in[2];
  const float* relinfo = (const float*)d_in[3];
  const float* rW1 = (const float*)d_in[4];
  const float* rb1 = (const float*)d_in[5];
  const float* rW2 = (const float*)d_in[6];
  const float* rb2 = (const float*)d_in[7];
  const float* rW3 = (const float*)d_in[8];
  const float* rb3 = (const float*)d_in[9];
  const float* oW1 = (const float*)d_in[10];
  const float* ob1 = (const float*)d_in[11];
  const float* oW2 = (const float*)d_in[12];
  const float* ob2 = (const float*)d_in[13];
  const float* oW3 = (const float*)d_in[14];
  const float* ob3 = (const float*)d_in[15];
  (void)in_sizes; (void)n_in; (void)out_size; (void)ws_size;

  char* ws = (char*)d_ws;
  size_t off = 0;
  auto alloc = [&](size_t bytes) -> void* {
    void* p = ws + off;
    off = (off + bytes + 255) & ~(size_t)255;
    return p;
  };
  bf16_t* W1p = (bf16_t*)alloc((size_t)1 * 10 * 512 * sizeof(bf16_t));
  bf16_t* W2p = (bf16_t*)alloc((size_t)5 * 10 * 512 * sizeof(bf16_t));
  bf16_t* W3p = (bf16_t*)alloc((size_t)5 * 7 * 512 * sizeof(bf16_t));
  bf16_t* U1p = (bf16_t*)alloc((size_t)4 * 7 * 512 * sizeof(bf16_t));
  bf16_t* U2p = (bf16_t*)alloc((size_t)4 * 7 * 512 * sizeof(bf16_t));
  bf16_t* U3p = (bf16_t*)alloc((size_t)4 * 1 * 512 * sizeof(bf16_t));
  bf16_t* X1  = (bf16_t*)alloc((size_t)B_SZ * RP * K1 * sizeof(bf16_t));
  bf16_t* Esw = (bf16_t*)alloc((size_t)B_SZ * KS_R * 7 * 512 * sizeof(bf16_t));
  float*  ERb = (float*)alloc((size_t)B_SZ * E_PAD * E_PAD * sizeof(float));

  prep_kernel<<<1, 256, 0, stream>>>(rW1, rb1, rW2, rb2, rW3, rb3,
                                     oW1, ob1, oW2, ob2, oW3, ob3,
                                     W1p, W2p, W3p, U1p, U2p, U3p);
  gather_kernel<<<dim3(RP / 256, B_SZ), 256, 0, stream>>>(objects, S, Rr, relinfo, X1);
  relmlp_kernel<<<dim3(RP / 128, B_SZ), 256, 0, stream>>>(X1, W1p, W2p, W3p, Esw);
  scatter_kernel<<<B_SZ, 256, 0, stream>>>(Rr, Esw, ERb);
  objmlp_kernel<<<B_SZ, 256, 0, stream>>>(objects, ERb, U1p, U2p, U3p, (float*)d_out);
}
